// Dual_Attention_layer_32246614458841
// MI455X (gfx1250) — matrix operand order repaired and checked by reading, not compiled, not run on hardware
//
#include <hip/hip_runtime.h>

// ---------------- problem constants ----------------
#define H_   8
#define DM_  128
#define N_   8192
#define B_   2
#define E_   131072
#define M2_  (B_ * N_)      // 16384 rows
#define CE_  16384          // edge chunk
#define SCALE_ 0.25f        // 1/sqrt(16)

typedef unsigned short u16;
typedef __attribute__((ext_vector_type(16))) __bf16 v16bf;
typedef __attribute__((ext_vector_type(8)))  float  v8f;

union Frag16 { v16bf v; u16 u[16]; uint4 q[2]; };

__device__ __forceinline__ u16 f2bf(float f) {
  unsigned x = __float_as_uint(f);
  x += 0x7FFFu + ((x >> 16) & 1u);          // round-to-nearest-even
  return (u16)(x >> 16);
}

// ======================================================================
// Generic GEMM: out[M x P] = act(A[M x K] @ W[K x P] + bias)
// bf16 WMMA, fp32 accumulate. Block = 128 thr (4 waves), block tile 64x64.
// W slab (K x 64) staged transposed into LDS as bf16, padded (ldw % 8 == 0
// keeps ds_load_b128 alignment; +8 pad rotates banks).
// Compile-time: K, ACT (0 none / 1 sigmoid / 2 relu), HASB (bias).
// ======================================================================
template <int K, int ACT, bool HASB>
__global__ __launch_bounds__(128) void gemm_bf16_kernel(
    const float* __restrict__ A, const float* __restrict__ W,
    const float* __restrict__ bias, float* __restrict__ out,
    int M, int P)
{
  extern __shared__ __align__(16) u16 wlds[];   // [64][K + 8] bf16, transposed
  constexpr int ldw = K + 8;
  const int colBase = blockIdx.y * 64;
  const int rowBase = blockIdx.x * 64;

  // ---- stage W[:, colBase .. colBase+64) transposed to LDS (fp32 -> bf16)
  for (int idx = threadIdx.x; idx < K * 64; idx += 128) {
    int c = idx & 63;
    int k = idx >> 6;
    wlds[c * ldw + k] = f2bf(W[(size_t)k * P + colBase + c]);
  }
  __syncthreads();

  const int wave = threadIdx.x >> 5;
  const int lane = threadIdx.x & 31;
  const int half = lane >> 4;     // 0: lanes 0-15, 1: lanes 16-31
  const int l16  = lane & 15;

  const int mRow = rowBase + wave * 16 + l16;       // A-fragment row for this lane
  const float* __restrict__ arow = A + (size_t)mRow * K;

  v8f acc[4];
#pragma unroll
  for (int nt = 0; nt < 4; ++nt)
#pragma unroll
    for (int r = 0; r < 8; ++r) acc[nt][r] = 0.0f;

#pragma unroll
  for (int kk = 0; kk < K; kk += 32) {
    // ---- A fragment (16x32 bf16): per ISA layout
    // u[0..7]  = A[m][kk + half*8 + j],  u[8..15] = A[m][kk+16 + half*8 + j]
    Frag16 af;
    {
      const float4* p0 = (const float4*)(arow + kk + half * 8);
      const float4* p1 = (const float4*)(arow + kk + 16 + half * 8);
      float4 x0 = p0[0], x1 = p0[1], y0 = p1[0], y1 = p1[1];
      af.u[0] = f2bf(x0.x);  af.u[1] = f2bf(x0.y);  af.u[2]  = f2bf(x0.z);  af.u[3]  = f2bf(x0.w);
      af.u[4] = f2bf(x1.x);  af.u[5] = f2bf(x1.y);  af.u[6]  = f2bf(x1.z);  af.u[7]  = f2bf(x1.w);
      af.u[8] = f2bf(y0.x);  af.u[9] = f2bf(y0.y);  af.u[10] = f2bf(y0.z);  af.u[11] = f2bf(y0.w);
      af.u[12]= f2bf(y1.x);  af.u[13]= f2bf(y1.y);  af.u[14] = f2bf(y1.z);  af.u[15] = f2bf(y1.w);
    }
#pragma unroll
    for (int nt = 0; nt < 4; ++nt) {
      // ---- B fragment (32x16 bf16): u[j] = W[kk + half*8 + j][col], u[8+j] = W[kk+16 + half*8 + j][col], col = nt*16 + l16
      Frag16 bf;
      const uint4* bp = (const uint4*)(wlds + (size_t)(nt * 16 + l16) * ldw + kk + half * 8);
      bf.q[0] = bp[0];
      bf.q[1] = bp[2];
      acc[nt] = __builtin_amdgcn_wmma_f32_16x16x32_bf16(
          false, af.v, false, bf.v, (short)0, acc[nt], false, false);
    }
  }

  // ---- epilogue: C/D layout: VGPR r -> row (half*8 + r), col = l16
#pragma unroll
  for (int nt = 0; nt < 4; ++nt) {
    int col = colBase + nt * 16 + l16;
    float bv = HASB ? bias[col] : 0.0f;
#pragma unroll
    for (int r = 0; r < 8; ++r) {
      int row = rowBase + wave * 16 + half * 8 + r;
      float v = acc[nt][r] + bv;
      if (ACT == 1)      v = 1.0f / (1.0f + __expf(-v));
      else if (ACT == 2) v = fmaxf(v, 0.0f);
      out[(size_t)row * P + col] = v;
    }
  }
}

// ======================================================================
// zero-fill
// ======================================================================
__global__ void fill0_kernel(float* __restrict__ p, long n) {
  long i = (long)blockIdx.x * blockDim.x + threadIdx.x;
  if (i < n) p[i] = 0.0f;
}

// ======================================================================
// linear-attention KV reduction: KV[b,h,k,v] = sum_n K[b,n,h,k]*V[b,n,h,v]
// ksum[b,h,k] = sum_n K[b,n,h,k].  grid = (B*H, nChunks), block = 256.
// ======================================================================
__global__ __launch_bounds__(256) void kv_reduce_kernel(
    const float* __restrict__ Km, const float* __restrict__ Vm,
    float* __restrict__ KV, float* __restrict__ KS, int Nn, int chunk)
{
  int bh = blockIdx.x;  int b = bh >> 3;  int h = bh & 7;
  int n0 = blockIdx.y * chunk;
  int t = threadIdx.x;  int k = t >> 4;  int v = t & 15;
  const float* kp = Km + ((size_t)b * Nn + n0) * DM_ + h * 16 + k;
  const float* vp = Vm + ((size_t)b * Nn + n0) * DM_ + h * 16 + v;
  float acc = 0.0f, ks = 0.0f;
  for (int n = 0; n < chunk; ++n) {
    float kv = kp[(size_t)n * DM_];
    acc += kv * vp[(size_t)n * DM_];
    ks  += kv;
  }
  atomicAdd(&KV[((size_t)bh * 16 + k) * 16 + v], acc);
  if (v == 0) atomicAdd(&KS[bh * 16 + k], ks);
}

// ======================================================================
// out[b,n,h,v] = (Q[b,n,h,:] . KV[b,h,:,v]) / (Q[b,n,h,:] . ksum[b,h,:] + 1e-8)
// ======================================================================
__global__ void linattn_kernel(const float* __restrict__ Q,
                               const float* __restrict__ KV,
                               const float* __restrict__ KS,
                               float* __restrict__ out, int Nn)
{
  long i = (long)blockIdx.x * blockDim.x + threadIdx.x;   // over B*N*H*16
  int  v  = (int)(i & 15);
  long t  = i >> 4;
  int  h  = (int)(t % H_);
  long bn = t / H_;
  int  b  = (int)(bn / Nn);
  const float* q  = Q  + bn * DM_ + h * 16;
  const float* kv = KV + ((size_t)(b * H_ + h) * 16) * 16;
  const float* ks = KS + (size_t)(b * H_ + h) * 16;
  float num = 0.0f, den = 0.0f;
#pragma unroll
  for (int k = 0; k < 16; ++k) {
    float qv = q[k];
    num += qv * kv[k * 16 + v];
    den += qv * ks[k];
  }
  out[bn * DM_ + h * 16 + v] = num / (den + 1e-8f);
}

// ======================================================================
// per-edge attention: score = SCALE * sum_d Q[src]*K[tgt]*Eh, a = exp(clip)
// scatter-add a*V[tgt] into msg[src], a into coeff[src]. One thread = (e,h).
// ======================================================================
__global__ __launch_bounds__(256) void edge_kernel(
    const float* __restrict__ Eh, const float* __restrict__ Q,
    const float* __restrict__ Kt, const float* __restrict__ V,
    const int* __restrict__ src, const int* __restrict__ tgt,
    int b, int e0, float* __restrict__ msg, float* __restrict__ coeff, int Nn)
{
  int i  = blockIdx.x * blockDim.x + threadIdx.x;   // over CE_*H_
  int h  = i & 7;
  int ec = i >> 3;
  int e  = e0 + ec;
  int s  = src[e];
  int tg = tgt[e];
  const float* qp = Q  + ((size_t)(b * Nn) + s ) * DM_ + h * 16;
  const float* kp = Kt + ((size_t)(b * Nn) + tg) * DM_ + h * 16;
  const float* vp = V  + ((size_t)(b * Nn) + tg) * DM_ + h * 16;
  const float* ep = Eh + (size_t)ec * DM_ + h * 16;
  float sc = 0.0f;
#pragma unroll
  for (int d = 0; d < 16; ++d) sc += qp[d] * kp[d] * ep[d];
  sc *= SCALE_;
  sc = fminf(fmaxf(sc, -5.0f), 5.0f);
  float a = __expf(sc);
  float* mp = msg + ((size_t)(b * Nn) + s) * DM_ + h * 16;
#pragma unroll
  for (int d = 0; d < 16; ++d) atomicAdd(&mp[d], a * vp[d]);
  atomicAdd(&coeff[((size_t)(b * Nn) + s) * H_ + h], a);
}

// out[b,n,h,v] = msg / (coeff + 1e-8)
__global__ void edge_norm_kernel(const float* __restrict__ msg,
                                 const float* __restrict__ coeff,
                                 float* __restrict__ out)
{
  long i  = (long)blockIdx.x * blockDim.x + threadIdx.x;  // over M2_*128
  long bn = i >> 7;
  int  h  = (int)((i >> 4) & 7);
  out[i] = msg[i] / (coeff[bn * H_ + h] + 1e-8f);
}

// ======================================================================
// LayerNorm over DM=128 with optional residual. One 128-thread block / row.
// ======================================================================
__global__ __launch_bounds__(128) void ln_kernel(
    const float* __restrict__ x, const float* __restrict__ res,
    const float* __restrict__ g, const float* __restrict__ beta,
    float* __restrict__ out)
{
  __shared__ float sb[128];
  size_t row = blockIdx.x;
  int t = threadIdx.x;
  float v = x[row * DM_ + t];
  if (res) v += res[row * DM_ + t];
  sb[t] = v; __syncthreads();
  for (int s = 64; s > 0; s >>= 1) { if (t < s) sb[t] += sb[t + s]; __syncthreads(); }
  float mean = sb[0] * (1.0f / 128.0f);
  __syncthreads();
  float d = v - mean;
  sb[t] = d * d; __syncthreads();
  for (int s = 64; s > 0; s >>= 1) { if (t < s) sb[t] += sb[t + s]; __syncthreads(); }
  float var = sb[0] * (1.0f / 128.0f);
  out[row * DM_ + t] = g[t] * d * rsqrtf(var + 1e-5f) + beta[t];
}

// concat along feature dim: out[r, 0:128]=a, out[r, 128:256]=b
__global__ void concat_kernel(const float* __restrict__ a,
                              const float* __restrict__ b,
                              float* __restrict__ out)
{
  long i = (long)blockIdx.x * blockDim.x + threadIdx.x;  // over M2_*256
  long r = i >> 8;
  int  c = (int)(i & 255);
  out[i] = (c < 128) ? a[r * 128 + c] : b[r * 128 + (c - 128)];
}

// ======================================================================
// host orchestration
// ======================================================================
struct SelfP  { const float *Wq, *Wk, *Wv, *fc, *g, *b; };
struct CrossP { const float *Wq, *Wk, *Wv, *fc, *g, *b, *We; };
struct FusP   { const float *W, *bW, *g, *b; };
struct FfnP   { const float *W1, *W2, *g, *b; };

// dispatch over the five (K, act, bias) combinations actually used
static inline void gemm(hipStream_t s, const float* A, const float* W,
                        const float* bias, float* out, int M, int K, int P, int act)
{
  dim3 grid(M / 64, P / 64);
  dim3 block(128);
  size_t lds = (size_t)64 * (K + 8) * sizeof(u16);
  if (K == 128 && act == 0)
    gemm_bf16_kernel<128, 0, false><<<grid, block, lds, s>>>(A, W, nullptr, out, M, P);
  else if (K == 128 && act == 1)
    gemm_bf16_kernel<128, 1, false><<<grid, block, lds, s>>>(A, W, nullptr, out, M, P);
  else if (K == 128 && act == 2)
    gemm_bf16_kernel<128, 2, false><<<grid, block, lds, s>>>(A, W, nullptr, out, M, P);
  else if (K == 256)
    gemm_bf16_kernel<256, 2, true><<<grid, block, lds, s>>>(A, W, bias, out, M, P);
  else // K == 512, act == 0
    gemm_bf16_kernel<512, 0, false><<<grid, block, lds, s>>>(A, W, nullptr, out, M, P);
}

static inline void fill0(hipStream_t s, float* p, long n)
{
  fill0_kernel<<<(unsigned)((n + 255) / 256), 256, 0, s>>>(p, n);
}

extern "C" void kernel_launch(void* const* d_in, const int* in_sizes, int n_in,
                              void* d_out, int out_size, void* d_ws, size_t ws_size,
                              hipStream_t stream)
{
  (void)in_sizes; (void)n_in; (void)out_size; (void)ws_size;

  const float* con = (const float*)d_in[0];
  const int*   ei  = (const int*)  d_in[1];
  const float* ef  = (const float*)d_in[2];
  const float* var = (const float*)d_in[3];

  // params in dict order (see reference setup_inputs)
  SelfP  saV  = { (const float*)d_in[4],  (const float*)d_in[5],  (const float*)d_in[6],
                  (const float*)d_in[7],  (const float*)d_in[8],  (const float*)d_in[9]  };
  SelfP  saC  = { (const float*)d_in[10], (const float*)d_in[11], (const float*)d_in[12],
                  (const float*)d_in[13], (const float*)d_in[14], (const float*)d_in[15] };
  CrossP caV2C= { (const float*)d_in[16], (const float*)d_in[17], (const float*)d_in[18],
                  (const float*)d_in[19], (const float*)d_in[20], (const float*)d_in[21],
                  (const float*)d_in[22] };
  CrossP caC2V= { (const float*)d_in[23], (const float*)d_in[24], (const float*)d_in[25],
                  (const float*)d_in[26], (const float*)d_in[27], (const float*)d_in[28],
                  (const float*)d_in[29] };
  FusP   fusC = { (const float*)d_in[30], (const float*)d_in[31],
                  (const float*)d_in[32], (const float*)d_in[33] };
  FusP   fusV = { (const float*)d_in[34], (const float*)d_in[35],
                  (const float*)d_in[36], (const float*)d_in[37] };
  FfnP   ffnC = { (const float*)d_in[38], (const float*)d_in[39],
                  (const float*)d_in[40], (const float*)d_in[41] };
  FfnP   ffnV = { (const float*)d_in[42], (const float*)d_in[43],
                  (const float*)d_in[44], (const float*)d_in[45] };

  // reference uses batch-0 indices for both batches
  const int* con_idx = ei;        // ei[0][0]
  const int* var_idx = ei + E_;   // ei[0][1]

  // -------- workspace layout (fp32 slots of M2_*128) --------
  float* ws = (float*)d_ws;
  const size_t S = (size_t)M2_ * DM_;      // 2,097,152 floats = 8 MB
  float* Qb  = ws;            // also comb_c later
  float* Kb  = ws + 1 * S;    // also comb_v later
  float* Vb  = ws + 2 * S;
  float* ATT = ws + 3 * S;
  float* PR  = ws + 4 * S;
  float* SV  = ws + 5 * S;
  float* SC  = ws + 6 * S;
  float* CV  = ws + 7 * S;
  float* CC  = ws + 8 * S;
  float* CAT = ws + 9 * S;    // 2 slots (M2_ x 256)
  float* HID = ws + 11 * S;   // 4 slots (M2_ x 512)
  float* EH  = ws + 15 * S;   // 1 slot (CE_ x 128)
  float* MSG = ws + 16 * S;   // 1 slot
  float* CO  = ws + 17 * S;               // M2_*H_ coeffs (contiguous after MSG)
  float* KV  = CO + (size_t)M2_ * H_;     // B*H*16*16
  float* KS  = KV + B_ * H_ * 256;        // B*H*16

  float* out_v = (float*)d_out;
  float* out_c = out_v + S;

  auto self_attn = [&](const float* x, const SelfP& p, float* dst) {
    gemm(stream, x, p.Wq, nullptr, Qb, M2_, 128, 128, 1);   // sigmoid
    gemm(stream, x, p.Wk, nullptr, Kb, M2_, 128, 128, 1);   // sigmoid
    gemm(stream, x, p.Wv, nullptr, Vb, M2_, 128, 128, 0);
    fill0(stream, KV, (long)B_ * H_ * 256 + B_ * H_ * 16);  // KV + KS contiguous
    kv_reduce_kernel<<<dim3(B_ * H_, 64), 256, 0, stream>>>(Kb, Vb, KV, KS, N_, N_ / 64);
    linattn_kernel<<<(M2_ * DM_) / 256, 256, 0, stream>>>(Qb, KV, KS, ATT, N_);
    gemm(stream, ATT, p.fc, nullptr, PR, M2_, 128, 128, 0);
    ln_kernel<<<M2_, 128, 0, stream>>>(PR, x, p.g, p.b, dst);
  };

  auto cross_attn = [&](const int* src, const int* tgt, const float* xq,
                        const float* xkv, const CrossP& p, float* dst) {
    gemm(stream, xq,  p.Wq, nullptr, Qb, M2_, 128, 128, 0);
    gemm(stream, xkv, p.Wk, nullptr, Kb, M2_, 128, 128, 0);
    gemm(stream, xkv, p.Wv, nullptr, Vb, M2_, 128, 128, 0);
    fill0(stream, MSG, (long)S + (long)M2_ * H_);           // MSG + CO contiguous
    for (int b = 0; b < B_; ++b) {
      for (int c = 0; c < E_ / CE_; ++c) {
        gemm(stream, ef + ((size_t)b * E_ + (size_t)c * CE_) * DM_, p.We,
             nullptr, EH, CE_, 128, 128, 0);
        edge_kernel<<<(CE_ * H_) / 256, 256, 0, stream>>>(
            EH, Qb, Kb, Vb, src, tgt, b, c * CE_, MSG, CO, N_);
      }
    }
    edge_norm_kernel<<<(M2_ * DM_) / 256, 256, 0, stream>>>(MSG, CO, ATT);
    gemm(stream, ATT, p.fc, nullptr, PR, M2_, 128, 128, 0);
    ln_kernel<<<M2_, 128, 0, stream>>>(PR, xq, p.g, p.b, dst);
  };

  auto fusion = [&](const float* a, const float* c, const FusP& p, float* dst) {
    concat_kernel<<<(M2_ * 256) / 256, 256, 0, stream>>>(a, c, CAT);
    gemm(stream, CAT, p.W, p.bW, PR, M2_, 256, 128, 2);     // +bias, relu
    ln_kernel<<<M2_, 128, 0, stream>>>(PR, nullptr, p.g, p.b, dst);
  };

  auto ffn = [&](const float* x, const FfnP& p, float* dst) {
    gemm(stream, x, p.W1, nullptr, HID, M2_, 128, 512, 2);  // relu
    gemm(stream, HID, p.W2, nullptr, PR, M2_, 512, 128, 0);
    ln_kernel<<<M2_, 128, 0, stream>>>(PR, x, p.g, p.b, dst);
  };

  // ---- full layer ----
  self_attn(var, saV, SV);
  self_attn(con, saC, SC);
  cross_attn(var_idx, con_idx, var, con, caV2C, CV);
  cross_attn(con_idx, var_idx, con, var, caC2V, CC);

  fusion(SC, CC, fusC, Qb);   // comb_c  (Q slot now free)
  fusion(SV, CV, fusV, Kb);   // comb_v

  ffn(Qb, ffnC, out_c);
  ffn(Kb, ffnV, out_v);
}
